// SqlParser_76209899700233
// MI455X (gfx1250) — matrix operand order repaired and checked by reading, not compiled, not run on hardware
//
#include <hip/hip_runtime.h>
#include <math.h>

// ---------------- constants ----------------
#define BB   256
#define SS   122
#define LQv  50
#define LTv  12
#define LCv  60
#define TT   80
#define DD   512
#define INDIM 1408
#define NHh  8
#define DHh  64
#define NPRODv 96   // prod_embed has NPROD+1 = 97 rows

typedef unsigned short u16;
typedef __attribute__((ext_vector_type(16))) __bf16 v16bf;
typedef __attribute__((ext_vector_type(8)))  float  v8f;

#define HAS_ASYNC_LDS (__has_builtin(__builtin_amdgcn_global_load_async_to_lds_b64))

#if HAS_ASYNC_LDS
typedef int v2i_ty __attribute__((vector_size(8)));
typedef __attribute__((address_space(1))) v2i_ty* g_v2i_p;
typedef __attribute__((address_space(3))) v2i_ty* l_v2i_p;
#endif

// ---------------- helpers ----------------
__device__ __forceinline__ u16 f2bf(float f) {
  unsigned x = __float_as_uint(f);
  unsigned r = x + 0x7fffu + ((x >> 16) & 1u);
  return (u16)(r >> 16);
}
__device__ __forceinline__ float bf2f(u16 u) { return __uint_as_float(((unsigned)u) << 16); }
__device__ __forceinline__ float sigmf(float x) { return 1.0f / (1.0f + __expf(-x)); }

// ---------------- conversion kernels ----------------
__global__ void k_f32_to_bf16(const float* __restrict__ src, u16* __restrict__ dst, long n) {
  long i = (long)blockIdx.x * blockDim.x + threadIdx.x;
  if (i < n) dst[i] = f2bf(src[i]);
}

// contiguous bf16 slice of (B,Stot,width) -> (B,rowsPer,width)
__global__ void k_slice_us(const u16* __restrict__ src, u16* __restrict__ dst,
                           int rowsPer, int rowOff, int Stot, int width) {
  long i = (long)blockIdx.x * blockDim.x + threadIdx.x;
  long n = (long)BB * rowsPer * width;
  if (i < n) {
    int w = (int)(i % width);
    long r = i / width;
    int b = (int)(r / rowsPer);
    int s = (int)(r % rowsPer);
    dst[i] = src[((long)b * Stot + rowOff + s) * width + w];
  }
}

// ---------------- B-operand swizzle into WMMA fragment layout ----------------
// N padded to a multiple of 64 (4 tiles) so the GEMM never needs a validity branch.
// tile (kt,nt) of 32x16 stored at ((nt*Kt)+kt)*512 u16; within tile:
// lane = (n&15) + 16*(kk>=16), elem = kk&15  -> per-lane contiguous 16 bf16
__device__ __forceinline__ void swz_store(u16* __restrict__ dst, int Kt, int k, int n, u16 v) {
  int kt = k >> 5, kk = k & 31;
  int nt = n >> 4, nn = n & 15;
  int lane = nn + ((kk >= 16) ? 16 : 0);
  int elem = kk & 15;
  dst[(((long)nt * Kt + kt) * 32 + lane) * 16 + elem] = v;
}

__global__ void k_swz_f32(const float* __restrict__ src, u16* __restrict__ dst,
                          int K, int N, int ldb) {
  int Kt = (K + 31) >> 5;
  int Kp = Kt * 32, Np = ((N + 63) >> 6) * 64;
  long total = (long)Kp * Np;
  long i = (long)blockIdx.x * blockDim.x + threadIdx.x;
  if (i >= total) return;
  int n = (int)(i / Kp);
  int k = (int)(i % Kp);
  u16 v = (k < K && n < N) ? f2bf(src[(long)k * ldb + n]) : (u16)0;
  swz_store(dst, Kt, k, n, v);
}

// B[k][n] = srcT[n*ldsrc + k]  (transposed source, f32)
__global__ void k_swz_f32T(const float* __restrict__ srcT, u16* __restrict__ dst,
                           int K, int N, int ldsrc) {
  int Kt = (K + 31) >> 5;
  int Kp = Kt * 32, Np = ((N + 63) >> 6) * 64;
  long total = (long)Kp * Np;
  long i = (long)blockIdx.x * blockDim.x + threadIdx.x;
  if (i >= total) return;
  int n = (int)(i / Kp);
  int k = (int)(i % Kp);
  u16 v = (k < K && n < N) ? f2bf(srcT[(long)n * ldsrc + k]) : (u16)0;
  swz_store(dst, Kt, k, n, v);
}

__global__ void k_swz_u16(const u16* __restrict__ src, u16* __restrict__ dst,
                          int K, int N, int ldb) {
  int Kt = (K + 31) >> 5;
  int Kp = Kt * 32, Np = ((N + 63) >> 6) * 64;
  long total = (long)Kp * Np;
  long i = (long)blockIdx.x * blockDim.x + threadIdx.x;
  if (i >= total) return;
  int n = (int)(i / Kp);
  int k = (int)(i % Kp);
  u16 v = (k < K && n < N) ? src[(long)k * ldb + n] : (u16)0;
  swz_store(dst, Kt, k, n, v);
}

// ---------------- WMMA GEMM: C[MxN] = A[MxK](bf16,row) @ Bsw(swizzled,padded) ----
// REQUIRES: M % 32 == 0, K % 32 == 0 (true for every call site). N arbitrary.
#define GF_BIAS 1
#define GF_ACC  2
#define GF_TANH 4

__global__ __launch_bounds__(256) void k_gemm_bf16(
    const u16* __restrict__ A, int lda,
    const u16* __restrict__ Bsw,
    float* __restrict__ Cf, u16* __restrict__ Cb, int ldc, int ldob,
    const float* __restrict__ bias,
    int M, int N, int K, int flags) {
  __shared__ u16 Asw[2][2][32][16];  // double-buffered fragment-layout A tile (4 KB)
  const int tid  = threadIdx.x;
  const int wave = tid >> 5;
  const int lane = tid & 31;
  const int wm = wave >> 2;   // 0..1
  const int wn = wave & 3;    // 0..3
  const int m0 = blockIdx.y * 32;
  const int half = lane >> 4;
  const int l15  = lane & 15;
  const int Kt = (K + 31) >> 5;
  const int ntile = blockIdx.x * 4 + wn;   // always backed (B padded to 4-tile multiple)
  const u16* bbase = Bsw + ((size_t)ntile * Kt) * 512 + (size_t)lane * 16;

  // A staging: all 256 threads, one aligned 8-byte (4 bf16) chunk each
  const int sm  = tid >> 3;                         // row in 32-row tile
  const int skg = tid & 7;                          // chunk-of-4 index
  const int s_wm   = sm >> 4;
  const int s_lane = (sm & 15) + 16 * (skg >> 2);
  const int s_elem = (skg & 3) * 4;
  const u16* arow = A + (size_t)(m0 + sm) * lda + skg * 4;

  union U { uint4 u[2]; v16bf v; };
  v8f acc = {};

  // prologue: stage k-tile 0
  {
    uint2 a0 = *(const uint2*)arow;
    *(uint2*)&Asw[0][s_wm][s_lane][s_elem] = a0;
  }
  __syncthreads();

  for (int kt = 0; kt < Kt; ++kt) {
    const int buf = kt & 1;
    const bool have_next = (kt + 1 < Kt);

#if HAS_ASYNC_LDS
    // CDNA5 async global->LDS staging of next A tile, tracked by ASYNCcnt so the
    // pre-WMMA LOADcnt wait only covers this tile's B fragments.
    if (have_next) {
      __builtin_amdgcn_global_load_async_to_lds_b64(
          (g_v2i_p)(arow + (size_t)(kt + 1) * 32),
          (l_v2i_p)&Asw[buf ^ 1][s_wm][s_lane][s_elem],
          0, 0);
    }
#else
    uint2 nxt;
    if (have_next) nxt = *(const uint2*)(arow + (size_t)(kt + 1) * 32);
#endif
    if (kt + 2 < Kt) __builtin_prefetch(bbase + (size_t)(kt + 2) * 512, 0, 1);

    U a, b;
    const uint4* bp = (const uint4*)(bbase + (size_t)kt * 512);
    b.u[0] = bp[0];
    b.u[1] = bp[1];
    a.u[0] = *(const uint4*)&Asw[buf][wm][lane][0];
    a.u[1] = *(const uint4*)&Asw[buf][wm][lane][8];
    acc = __builtin_amdgcn_wmma_f32_16x16x32_bf16(false, a.v, false, b.v, (short)0, acc, false, false);

#if HAS_ASYNC_LDS
#if __has_builtin(__builtin_amdgcn_s_wait_asynccnt)
    __builtin_amdgcn_s_wait_asynccnt(0);
#else
    asm volatile("s_wait_asynccnt 0x0" ::: "memory");
#endif
#else
    if (have_next) *(uint2*)&Asw[buf ^ 1][s_wm][s_lane][s_elem] = nxt;
#endif
    __syncthreads();
  }

  const int n = ntile * 16 + l15;
  if (n < N) {
#pragma unroll
    for (int r = 0; r < 8; ++r) {  // C: lane->N, vgpr r -> M=r (+8 for upper lane half)
      int m = m0 + wm * 16 + r + half * 8;
      float v = acc[r];
      if (flags & GF_BIAS) v += bias[n];
      if (flags & GF_ACC)  v += Cf[(long)m * ldc + n];
      if (flags & GF_TANH) v = tanhf(v);
      if (Cf) Cf[(long)m * ldc + n] = v;
      if (Cb) Cb[(long)m * ldob + n] = f2bf(v);
    }
  }
}

// ---------------- vectorized bf16 dot helper ----------------
__device__ __forceinline__ float dot64_bf16(const float* __restrict__ qs, const u16* __restrict__ kr) {
  const uint4* kr4 = (const uint4*)kr;
  float d = 0.f;
#pragma unroll
  for (int i = 0; i < 8; ++i) {
    uint4 p = kr4[i];
    d += qs[i * 8 + 0] * bf2f((u16)(p.x & 0xffff));
    d += qs[i * 8 + 1] * bf2f((u16)(p.x >> 16));
    d += qs[i * 8 + 2] * bf2f((u16)(p.y & 0xffff));
    d += qs[i * 8 + 3] * bf2f((u16)(p.y >> 16));
    d += qs[i * 8 + 4] * bf2f((u16)(p.z & 0xffff));
    d += qs[i * 8 + 5] * bf2f((u16)(p.z >> 16));
    d += qs[i * 8 + 6] * bf2f((u16)(p.w & 0xffff));
    d += qs[i * 8 + 7] * bf2f((u16)(p.w >> 16));
  }
  return d;
}

// ---------------- attention with context output (enc attention) ----------------
__global__ __launch_bounds__(128) void k_attn_ctx(
    const float* __restrict__ Q, const u16* __restrict__ Kb, const u16* __restrict__ Vb,
    const unsigned char* __restrict__ mask, u16* __restrict__ Ob, int S) {
  __shared__ float qs[64];
  __shared__ float sc[128];
  __shared__ float red[128];
  const int b = blockIdx.x >> 3, h = blockIdx.x & 7, tid = threadIdx.x;
  if (tid < 64) qs[tid] = Q[(long)b * DD + h * DHh + tid];
  __syncthreads();
  float score = -1e30f;
  if (tid < S) {
    float d = dot64_bf16(qs, Kb + ((long)b * S + tid) * DD + h * DHh) * 0.125f;
    if (!mask[(long)b * S + tid]) d = -1e9f;
    score = d;
  }
  red[tid] = score; __syncthreads();
  for (int s = 64; s > 0; s >>= 1) { if (tid < s) red[tid] = fmaxf(red[tid], red[tid + s]); __syncthreads(); }
  float mx = red[0]; __syncthreads();
  float e = (tid < S) ? __expf(score - mx) : 0.f;
  sc[tid] = e; red[tid] = e; __syncthreads();
  for (int s = 64; s > 0; s >>= 1) { if (tid < s) red[tid] += red[tid + s]; __syncthreads(); }
  float Z = red[0]; __syncthreads();
  if (tid < 64) {
    float o = 0.f;
    for (int s = 0; s < S; ++s) o += sc[s] * bf2f(Vb[((long)b * S + s) * DD + h * DHh + tid]);
    Ob[(long)b * DD + h * DHh + tid] = f2bf(o / Z);
  }
}

// ---------------- prob-only attention (col/tab) ----------------
__global__ __launch_bounds__(64) void k_attn_prob(
    const float* __restrict__ Q, const u16* __restrict__ Kb,
    const unsigned char* __restrict__ mask, int maskOff, int maskStride,
    float* __restrict__ P, int Sm) {
  __shared__ float qs[64];
  __shared__ float red[64];
  const int b = blockIdx.x >> 3, h = blockIdx.x & 7, tid = threadIdx.x;
  qs[tid] = Q[(long)b * DD + h * DHh + tid];
  __syncthreads();
  float score = -1e30f, e = 0.f;
  if (tid < Sm) {
    float d = dot64_bf16(qs, Kb + ((long)b * Sm + tid) * DD + h * DHh) * 0.125f;
    if (!mask[(long)b * maskStride + maskOff + tid]) d = -1e9f;
    score = d;
  }
  red[tid] = score; __syncthreads();
  for (int s = 32; s > 0; s >>= 1) { if (tid < s) red[tid] = fmaxf(red[tid], red[tid + s]); __syncthreads(); }
  float mx = red[0]; __syncthreads();
  e = (tid < Sm) ? __expf(score - mx) : 0.f;
  red[tid] = e; __syncthreads();
  for (int s = 32; s > 0; s >>= 1) { if (tid < s) red[tid] += red[tid + s]; __syncthreads(); }
  float Z = red[0];
  if (tid < Sm) P[((long)b * NHh + h) * Sm + tid] = e / Z;
}

__global__ void k_prob_mean_log(const float* __restrict__ P, float* __restrict__ lp, int Sm) {
  int i = blockIdx.x * blockDim.x + threadIdx.x;
  if (i < BB * Sm) {
    int b = i / Sm, s = i % Sm;
    float a = 0.f;
    for (int h = 0; h < NHh; ++h) a += P[((long)b * NHh + h) * Sm + s];
    lp[i] = __logf(a * 0.125f + 1e-32f);
  }
}

// ---------------- rule log-softmax (N=97) ----------------
__global__ __launch_bounds__(128) void k_rule_lp(const float* __restrict__ logits,
                                                 float* __restrict__ lp, int N) {
  __shared__ float red[128];
  const int b = blockIdx.x, tid = threadIdx.x;
  float v = (tid < N) ? logits[(long)b * N + tid] : -1e30f;
  red[tid] = v; __syncthreads();
  for (int s = 64; s > 0; s >>= 1) { if (tid < s) red[tid] = fmaxf(red[tid], red[tid + s]); __syncthreads(); }
  float mx = red[0]; __syncthreads();
  float e = (tid < N) ? __expf(v - mx) : 0.f;
  red[tid] = e; __syncthreads();
  for (int s = 64; s > 0; s >>= 1) { if (tid < s) red[tid] += red[tid + s]; __syncthreads(); }
  float lse = mx + __logf(red[0]);
  if (tid < N) lp[(long)b * N + tid] = v - lse;
}

// ---------------- LSTM gates ----------------
__global__ void k_lstm_gates(const float* __restrict__ g, float* __restrict__ c,
                             u16* __restrict__ cat, u16* __restrict__ hist, int t) {
  int i = blockIdx.x * blockDim.x + threadIdx.x;
  if (i < BB * DD) {
    int b = i >> 9, j = i & 511;
    const float* gr = g + (long)b * 2048;
    float gi = gr[j], gf = gr[512 + j], gg = gr[1024 + j], go = gr[1536 + j];
    float c2 = sigmf(gf) * c[i] + sigmf(gi) * tanhf(gg);
    float h2 = sigmf(go) * tanhf(c2);
    c[i] = c2;
    u16 hb = f2bf(h2);
    cat[(long)b * 1024 + j] = hb;
    hist[((long)t * BB + b) * DD + j] = hb;
  }
}

__global__ void k_init_state(const float* __restrict__ h0, float* __restrict__ c,
                             u16* __restrict__ cat) {
  int i = blockIdx.x * blockDim.x + threadIdx.x;
  if (i < BB * DD) {
    int b = i >> 9, j = i & 511;
    c[i] = 0.f;
    cat[(long)b * 1024 + j] = f2bf(h0[i]);
  }
}

// ---------------- build x (bf16, 1408) ----------------
__global__ __launch_bounds__(256) void k_build_x(
    u16* __restrict__ xb, const float* __restrict__ ctx, const u16* __restrict__ hist,
    const float* __restrict__ colE, const float* __restrict__ tabE,
    const float* __restrict__ prodE, const float* __restrict__ fieldE, const float* __restrict__ typeE,
    const int* __restrict__ act_type, const int* __restrict__ act_idx,
    const int* __restrict__ action_len, const int* __restrict__ parent_t,
    const int* __restrict__ f_prod, const int* __restrict__ f_field, const int* __restrict__ f_type,
    int t) {
  const int b = blockIdx.x;
  const int valid = (t < action_len[b]) ? 1 : 0;
  int ptype = 0, pidx = 0, fp = 0, ff = 0, ft = 0, pareff = 0;
  if (t > 0) {
    ptype = act_type[(long)(t - 1) * BB + b];
    pidx  = act_idx[(long)(t - 1) * BB + b];
    fp = f_prod[(long)t * BB + b];
    ff = f_field[(long)t * BB + b];
    ft = f_type[(long)t * BB + b];
    int par = parent_t[(long)t * BB + b];
    int cap = t - 1;
    pareff = valid ? ((par < cap) ? par : cap) : 0;
  }
  u16* xr = xb + (long)b * INDIM;
  for (int d = threadIdx.x; d < INDIM; d += 256) {
    float v = 0.f;
    if (d < 128) {
      if (t > 0 && valid) {
        if (ptype <= 1)      v = prodE[(long)((ptype == 1) ? NPRODv : pidx) * 128 + d];
        else if (ptype == 2) v = colE[((long)b * LCv + pidx) * 128 + d];
        else                 v = tabE[((long)b * LTv + pidx) * 128 + d];
      }
    } else if (d < 640) {
      v = ctx[(long)b * DD + (d - 128)];
    } else if (d < 768) {
      if (t > 0) v = prodE[(long)fp * 128 + (d - 640)];
    } else if (d < 832) {
      if (t > 0) v = fieldE[(long)ff * 64 + (d - 768)];
    } else if (d < 896) {
      v = (t > 0) ? typeE[(long)ft * 64 + (d - 832)] : typeE[d - 832];
    } else {
      if (t > 0) { xr[d] = hist[((long)pareff * BB + b) * DD + (d - 896)]; continue; }
    }
    xr[d] = f2bf(v);
  }
}

// ---------------- loss gather + reductions ----------------
__global__ __launch_bounds__(256) void k_gather_loss(
    const float* __restrict__ rule_lp, const float* __restrict__ col_lp, const float* __restrict__ tab_lp,
    const int* __restrict__ act_type, const int* __restrict__ act_idx, const int* __restrict__ action_len,
    float* __restrict__ loss_partial, int t) {
  __shared__ float red[256];
  const int b = threadIdx.x;
  int atype = act_type[(long)t * BB + b];
  int aidx  = act_idx[(long)t * BB + b];
  float lp;
  if (atype == 0)      lp = rule_lp[(long)b * 97 + aidx];
  else if (atype == 1) lp = rule_lp[(long)b * 97 + NPRODv];
  else if (atype == 2) lp = col_lp[(long)b * LCv + aidx];
  else                 lp = tab_lp[(long)b * LTv + aidx];
  red[b] = (t < action_len[b]) ? lp : 0.f;
  __syncthreads();
  for (int s = 128; s > 0; s >>= 1) { if (b < s) red[b] += red[b + s]; __syncthreads(); }
  if (b == 0) loss_partial[t] = red[0];
}

__global__ __launch_bounds__(128) void k_final(const float* __restrict__ loss_partial,
                                               float* __restrict__ out) {
  __shared__ float red[128];
  int i = threadIdx.x;
  red[i] = (i < TT) ? loss_partial[i] : 0.f;
  __syncthreads();
  for (int s = 64; s > 0; s >>= 1) { if (i < s) red[i] += red[i + s]; __syncthreads(); }
  if (i == 0) out[0] = -red[0];
}

// ---------------- host ----------------
static inline size_t swbytes(int K, int N) {
  return (size_t)((K + 31) / 32) * (((N + 63) / 64) * 4) * 512 * 2;
}

extern "C" void kernel_launch(void* const* d_in, const int* in_sizes, int n_in,
                              void* d_out, int out_size, void* d_ws, size_t ws_size,
                              hipStream_t stream) {
  (void)in_sizes; (void)n_in; (void)out_size; (void)ws_size;
  const float* enc        = (const float*)d_in[0];
  const unsigned char* mask = (const unsigned char*)d_in[1];
  const float* h0         = (const float*)d_in[2];
  const int* act_type     = (const int*)d_in[3];
  const int* act_idx      = (const int*)d_in[4];
  const int* action_len   = (const int*)d_in[5];
  const int* parent_t     = (const int*)d_in[6];
  const int* f_prod       = (const int*)d_in[7];
  const int* f_field      = (const int*)d_in[8];
  const int* f_type       = (const int*)d_in[9];
  const float* prodE      = (const float*)d_in[10];
  const float* fieldE     = (const float*)d_in[11];
  const float* typeE      = (const float*)d_in[12];
  const float* col_in_W   = (const float*)d_in[13];
  const float* col_in_b   = (const float*)d_in[14];
  const float* lstm_Wi    = (const float*)d_in[15];
  const float* lstm_Wh    = (const float*)d_in[16];
  const float* lstm_b     = (const float*)d_in[17];
  const float* ctx_Wq     = (const float*)d_in[18];
  const float* ctx_Wk     = (const float*)d_in[19];
  const float* ctx_Wv     = (const float*)d_in[20];
  const float* ctx_Wo     = (const float*)d_in[21];
  const float* col_Wq     = (const float*)d_in[22];
  const float* col_Wk     = (const float*)d_in[23];
  const float* tab_Wq     = (const float*)d_in[26];
  const float* tab_Wk     = (const float*)d_in[27];
  const float* att_W      = (const float*)d_in[30];
  const float* att_b      = (const float*)d_in[31];
  const float* rule_W     = (const float*)d_in[32];

  char* base = (char*)d_ws;
  size_t off = 0;
  auto alloc = [&](size_t bytes) -> void* {
    void* r = base + off;
    off += (bytes + 255) & ~(size_t)255;
    return r;
  };

  // row-major bf16 A-side statics
  u16* enc_b    = (u16*)alloc((size_t)BB * SS * DD * 2);
  u16* colenc_b = (u16*)alloc((size_t)BB * LCv * DD * 2);
  u16* tabenc_b = (u16*)alloc((size_t)BB * LTv * DD * 2);
  u16* ruleW_b  = (u16*)alloc((size_t)DD * 128 * 2);
  // swizzled B-side weights (N padded to 64)
  u16* Wi_sw    = (u16*)alloc(swbytes(INDIM, 2048));
  u16* Wh_sw    = (u16*)alloc(swbytes(DD, 2048));
  u16* ctxWq_sw = (u16*)alloc(swbytes(DD, DD));
  u16* ctxWk_sw = (u16*)alloc(swbytes(DD, DD));
  u16* ctxWv_sw = (u16*)alloc(swbytes(DD, DD));
  u16* ctxWo_sw = (u16*)alloc(swbytes(DD, DD));
  u16* colWq_sw = (u16*)alloc(swbytes(DD, DD));
  u16* colWk_sw = (u16*)alloc(swbytes(DD, DD));
  u16* tabWq_sw = (u16*)alloc(swbytes(DD, DD));
  u16* tabWk_sw = (u16*)alloc(swbytes(DD, DD));
  u16* attW_sw  = (u16*)alloc(swbytes(1024, DD));
  u16* colinW_sw= (u16*)alloc(swbytes(DD, 128));
  u16* prodT_sw = (u16*)alloc(swbytes(128, 97));
  u16* R_sw     = (u16*)alloc(swbytes(DD, 97));
  u16* R_rm     = (u16*)alloc((size_t)DD * 97 * 2);
  // precomputed K/V + heads (attention-side, row-major bf16)
  u16* encK_b = (u16*)alloc((size_t)BB * SS * DD * 2);
  u16* encV_b = (u16*)alloc((size_t)BB * SS * DD * 2);
  u16* colK_b = (u16*)alloc((size_t)BB * LCv * DD * 2);
  u16* tabK_b = (u16*)alloc((size_t)BB * LTv * DD * 2);
  float* colE = (float*)alloc((size_t)BB * LCv * 128 * 4);
  float* tabE = (float*)alloc((size_t)BB * LTv * 128 * 4);
  // recurrent state
  u16* hist_b = (u16*)alloc((size_t)TT * BB * DD * 2);
  u16* x_b    = (u16*)alloc((size_t)BB * INDIM * 2);
  float* g    = (float*)alloc((size_t)BB * 2048 * 4);
  float* c    = (float*)alloc((size_t)BB * DD * 4);
  u16* cat_b  = (u16*)alloc((size_t)BB * 1024 * 2);   // [h2 | ctx] bf16
  float* qbuf = (float*)alloc((size_t)BB * DD * 4);
  float* ctx  = (float*)alloc((size_t)BB * DD * 4);
  u16* ctxh_b = (u16*)alloc((size_t)BB * DD * 2);
  u16* att_bb = (u16*)alloc((size_t)BB * DD * 2);
  float* rlog = (float*)alloc((size_t)BB * 97 * 4);
  float* rlp  = (float*)alloc((size_t)BB * 97 * 4);
  float* qcol = (float*)alloc((size_t)BB * DD * 4);
  float* qtab = (float*)alloc((size_t)BB * DD * 4);
  float* colP = (float*)alloc((size_t)BB * NHh * LCv * 4);
  float* tabP = (float*)alloc((size_t)BB * NHh * LTv * 4);
  float* col_lp = (float*)alloc((size_t)BB * LCv * 4);
  float* tab_lp = (float*)alloc((size_t)BB * LTv * 4);
  float* loss_partial = (float*)alloc((size_t)TT * 4);

  auto cvt = [&](const float* s, u16* d, long n) {
    k_f32_to_bf16<<<dim3((unsigned)((n + 255) / 256)), dim3(256), 0, stream>>>(s, d, n);
  };
  auto swzF = [&](const float* s, u16* d, int K, int N) {
    long total = (long)(((K + 31) / 32) * 32) * (((N + 63) / 64) * 64);
    k_swz_f32<<<dim3((unsigned)((total + 255) / 256)), dim3(256), 0, stream>>>(s, d, K, N, N);
  };
  auto gemm = [&](const u16* A, int lda, const u16* Bsw, float* Cf, u16* Cb,
                  int ldc, int ldob, const float* bias, int M, int N, int K, int flags) {
    dim3 grid((N + 63) / 64, (M + 31) / 32);
    k_gemm_bf16<<<grid, dim3(256), 0, stream>>>(A, lda, Bsw, Cf, Cb, ldc, ldob, bias, M, N, K, flags);
  };

  // ---- static conversions / swizzles ----
  cvt(enc, enc_b, (long)BB * SS * DD);
  {
    long n1 = (long)BB * LCv * DD, n2 = (long)BB * LTv * DD;
    k_slice_us<<<dim3((unsigned)((n1 + 255) / 256)), dim3(256), 0, stream>>>(enc_b, colenc_b, LCv, LQv + LTv, SS, DD);
    k_slice_us<<<dim3((unsigned)((n2 + 255) / 256)), dim3(256), 0, stream>>>(enc_b, tabenc_b, LTv, LQv, SS, DD);
  }
  cvt(rule_W, ruleW_b, (long)DD * 128);
  swzF(lstm_Wi, Wi_sw, INDIM, 2048);
  swzF(lstm_Wh, Wh_sw, DD, 2048);
  swzF(ctx_Wq, ctxWq_sw, DD, DD); swzF(ctx_Wk, ctxWk_sw, DD, DD);
  swzF(ctx_Wv, ctxWv_sw, DD, DD); swzF(ctx_Wo, ctxWo_sw, DD, DD);
  swzF(col_Wq, colWq_sw, DD, DD); swzF(col_Wk, colWk_sw, DD, DD);
  swzF(tab_Wq, tabWq_sw, DD, DD); swzF(tab_Wk, tabWk_sw, DD, DD);
  swzF(att_W, attW_sw, 1024, DD);
  swzF(col_in_W, colinW_sw, DD, 128);
  {
    long total = (long)128 * (((97 + 63) / 64) * 64);
    k_swz_f32T<<<dim3((unsigned)((total + 255) / 256)), dim3(256), 0, stream>>>(prodE, prodT_sw, 128, 97, 128);
  }

  // ---- static GEMMs (time-invariant K/V, col/tab embeds, folded rule matrix) ----
  gemm(enc_b, DD, ctxWk_sw, nullptr, encK_b, 0, DD, nullptr, BB * SS, DD, DD, 0);
  gemm(enc_b, DD, ctxWv_sw, nullptr, encV_b, 0, DD, nullptr, BB * SS, DD, DD, 0);
  gemm(colenc_b, DD, colWk_sw, nullptr, colK_b, 0, DD, nullptr, BB * LCv, DD, DD, 0);
  gemm(tabenc_b, DD, tabWk_sw, nullptr, tabK_b, 0, DD, nullptr, BB * LTv, DD, DD, 0);
  gemm(colenc_b, DD, colinW_sw, colE, nullptr, 128, 0, col_in_b, BB * LCv, 128, DD, GF_BIAS);
  gemm(tabenc_b, DD, colinW_sw, tabE, nullptr, 128, 0, col_in_b, BB * LTv, 128, DD, GF_BIAS);
  gemm(ruleW_b, 128, prodT_sw, nullptr, R_rm, 0, 97, nullptr, DD, 97, 128, 0);
  {
    long total = (long)(((DD + 31) / 32) * 32) * (((97 + 63) / 64) * 64);
    k_swz_u16<<<dim3((unsigned)((total + 255) / 256)), dim3(256), 0, stream>>>(R_rm, R_sw, DD, 97, 97);
  }

  // ---- init state + ctx0 from h0 ----
  k_init_state<<<dim3((BB * DD + 255) / 256), dim3(256), 0, stream>>>(h0, c, cat_b);
  gemm(cat_b, 1024, ctxWq_sw, qbuf, nullptr, DD, 0, nullptr, BB, DD, DD, 0);
  k_attn_ctx<<<dim3(BB * NHh), dim3(128), 0, stream>>>(qbuf, encK_b, encV_b, mask, ctxh_b, SS);
  gemm(ctxh_b, DD, ctxWo_sw, ctx, cat_b + DD, DD, 1024, nullptr, BB, DD, DD, 0);

  // ---- 80 decode steps ----
  for (int t = 0; t < TT; ++t) {
    k_build_x<<<dim3(BB), dim3(256), 0, stream>>>(x_b, ctx, hist_b, colE, tabE, prodE, fieldE, typeE,
                                                  act_type, act_idx, action_len, parent_t,
                                                  f_prod, f_field, f_type, t);
    gemm(x_b, INDIM, Wi_sw, g, nullptr, 2048, 0, lstm_b, BB, 2048, INDIM, GF_BIAS);
    gemm(cat_b, 1024, Wh_sw, g, nullptr, 2048, 0, nullptr, BB, 2048, DD, GF_ACC);
    k_lstm_gates<<<dim3((BB * DD + 255) / 256), dim3(256), 0, stream>>>(g, c, cat_b, hist_b, t);
    gemm(cat_b, 1024, ctxWq_sw, qbuf, nullptr, DD, 0, nullptr, BB, DD, DD, 0);
    k_attn_ctx<<<dim3(BB * NHh), dim3(128), 0, stream>>>(qbuf, encK_b, encV_b, mask, ctxh_b, SS);
    gemm(ctxh_b, DD, ctxWo_sw, ctx, cat_b + DD, DD, 1024, nullptr, BB, DD, DD, 0);
    gemm(cat_b, 1024, attW_sw, nullptr, att_bb, 0, DD, att_b, BB, DD, 1024, GF_BIAS | GF_TANH);
    gemm(att_bb, DD, R_sw, rlog, nullptr, 97, 0, nullptr, BB, 97, DD, 0);
    k_rule_lp<<<dim3(BB), dim3(128), 0, stream>>>(rlog, rlp, 97);
    gemm(att_bb, DD, colWq_sw, qcol, nullptr, DD, 0, nullptr, BB, DD, DD, 0);
    gemm(att_bb, DD, tabWq_sw, qtab, nullptr, DD, 0, nullptr, BB, DD, DD, 0);
    k_attn_prob<<<dim3(BB * NHh), dim3(64), 0, stream>>>(qcol, colK_b, mask, LQv + LTv, SS, colP, LCv);
    k_prob_mean_log<<<dim3((BB * LCv + 255) / 256), dim3(256), 0, stream>>>(colP, col_lp, LCv);
    k_attn_prob<<<dim3(BB * NHh), dim3(64), 0, stream>>>(qtab, tabK_b, mask, LQv, SS, tabP, LTv);
    k_prob_mean_log<<<dim3((BB * LTv + 255) / 256), dim3(256), 0, stream>>>(tabP, tab_lp, LTv);
    k_gather_loss<<<dim3(1), dim3(256), 0, stream>>>(rlp, col_lp, tab_lp, act_type, act_idx,
                                                     action_len, loss_partial, t);
  }
  k_final<<<dim3(1), dim3(128), 0, stream>>>(loss_partial, (float*)d_out);
}
